// GATLayer_51668456571283
// MI455X (gfx1250) — matrix operand order repaired and checked by reading, not compiled, not run on hardware
//
#include <hip/hip_runtime.h>

typedef __attribute__((ext_vector_type(16))) __bf16 v16bf;
typedef __attribute__((ext_vector_type(8)))  float  v8f;
typedef unsigned int tdm_v4u __attribute__((ext_vector_type(4)));
typedef int          tdm_v8i __attribute__((ext_vector_type(8)));
typedef int          tdm_v4i __attribute__((ext_vector_type(4)));

#define B_N  4096
#define FIN  256
#define NH   4
#define HD   64
#define HDIM 256   // NH*HD
#define NJ   (B_N / 32)

__device__ __forceinline__ unsigned short f32_to_bf16(float f) {
    union { float f; unsigned int u; } c; c.f = f;
    unsigned int u = c.u;
    u += 0x7FFFu + ((u >> 16) & 1u);   // round-to-nearest-even
    return (unsigned short)(u >> 16);
}

union V16 {
    v16bf v;
    uint4 q[2];
    unsigned short s[16];
};

// Issue a 2D TDM tile load: tensor is [tensor_h][tensor_w] bf16 (row stride
// stride_elems), tile is [tile_h][tile_w] starting at gaddr, packed row-major
// into LDS at lds_off. Tracked by TENSORcnt.
__device__ __forceinline__ void tdm_load_2d_bf16(unsigned lds_off,
                                                 unsigned long long gaddr,
                                                 unsigned tensor_w, unsigned tensor_h,
                                                 unsigned tile_w,   unsigned tile_h,
                                                 unsigned long long stride_elems)
{
    tdm_v4u g0;
    g0[0] = 1u;                                            // count=1 (valid user D#)
    g0[1] = lds_off;                                       // lds_addr (bytes)
    g0[2] = (unsigned)(gaddr & 0xFFFFFFFFull);             // global_addr[31:0]
    g0[3] = (unsigned)((gaddr >> 32) & 0x01FFFFFFull)      // global_addr[56:32]
          | (2u << 30);                                    // type = 2 ("image")
    tdm_v8i g1;
    g1[0] = (int)(1u << 16);                               // data_size=1 -> 2 bytes
    g1[1] = (int)((tensor_w & 0xFFFFu) << 16);             // tensor_dim0[15:0]
    g1[2] = (int)((tensor_w >> 16) | ((tensor_h & 0xFFFFu) << 16)); // dim0 hi | dim1 lo
    g1[3] = (int)((tensor_h >> 16) | (tile_w << 16));      // dim1 hi | tile_dim0
    g1[4] = (int)tile_h;                                   // tile_dim1 | tile_dim2=0
    g1[5] = (int)(stride_elems & 0xFFFFFFFFull);           // dim0_stride[31:0]
    g1[6] = (int)((stride_elems >> 32) & 0xFFFFull);       // dim0_stride[47:32]
    g1[7] = 0;
    tdm_v4i z4 = {0, 0, 0, 0};
    tdm_v8i z8 = {0, 0, 0, 0, 0, 0, 0, 0};
    __builtin_amdgcn_tensor_load_to_lds(g0, g1, z4, z4, z8, 0);
}

// ---------------- K0: f32 -> bf16 conversion ----------------
__global__ void k_cvt_bf16(const float* __restrict__ src,
                           unsigned short* __restrict__ dst, int n) {
    int i = blockIdx.x * blockDim.x + threadIdx.x;
    if (i < n) dst[i] = f32_to_bf16(src[i]);
}

// ---------------- K1: h = x @ W^T via WMMA bf16 ----------------
// grid (64, 4), block 128 (4 waves). Wave tile 16x64, WG tile 64x64.
// Writes h as f32 row-major AND h^T as bf16 [NH][HD][B] (vectorized b128
// stores: a C-tile lane holds one column over 8 consecutive rows).
__global__ void __launch_bounds__(128)
k_proj_wmma(const unsigned short* __restrict__ xb,   // [B, FIN] bf16
            const unsigned short* __restrict__ wb,   // [HDIM, FIN] bf16
            float* __restrict__ hf,                  // [B, HDIM] f32
            unsigned short* __restrict__ hTg)        // [NH][HD][B] bf16
{
    const int lane = threadIdx.x & 31;
    const int wv   = threadIdx.x >> 5;
    const int lm   = lane & 15;
    const bool hi  = lane >= 16;
    const int i0 = blockIdx.x * 64 + wv * 16;
    const int n0 = blockIdx.y * 64;
    const int hh = blockIdx.y;                       // 64-column block == head

    v8f acc[4] = {};
    const unsigned short* arowp = xb + (i0 + lm) * FIN;

    for (int k0 = 0; k0 < FIN; k0 += 32) {
        // A: lane<16 gets K 0-7 & 16-23; lane>=16 gets K 8-15 & 24-31
        V16 a;
        const unsigned short* ap = arowp + k0 + (hi ? 8 : 0);
        a.q[0] = *(const uint4*)ap;
        a.q[1] = *(const uint4*)(ap + 16);
#pragma unroll
        for (int nt = 0; nt < 4; ++nt) {
            // B[k][n] = W[n][k]: lane<16 -> K 0-7 & 16-23 of column n0+nt*16+lm
            V16 b;
            const unsigned short* bp =
                wb + (n0 + nt * 16 + lm) * FIN + k0 + (hi ? 8 : 0);
            b.q[0] = *(const uint4*)bp;
            b.q[1] = *(const uint4*)(bp + 16);
            acc[nt] = __builtin_amdgcn_wmma_f32_16x16x32_bf16(
                false, a.v, false, b.v, (short)0, acc[nt], false, false);
        }
    }
#pragma unroll
    for (int nt = 0; nt < 4; ++nt) {
        const int dh = nt * 16 + lm;                 // d within head (lane's column)
        uint4 pk;
        unsigned short* ps = (unsigned short*)&pk;
#pragma unroll
        for (int v = 0; v < 8; ++v) {
            int m = v + (hi ? 8 : 0);                // C layout: lanes>=16 hold M=v+8
            float val = acc[nt][v];
            hf[(i0 + m) * HDIM + n0 + nt * 16 + lm] = val;
            ps[v] = f32_to_bf16(val);                // consecutive v -> consecutive j
        }
        *(uint4*)(hTg + (size_t)hh * HD * B_N + (size_t)dh * B_N
                      + i0 + (hi ? 8 : 0)) = pk;
    }
}

// ---------------- K2: s = h . a_src, t = h . a_dst ----------------
__global__ void k_attn_coef(const float* __restrict__ hf,
                            const float* __restrict__ a_src,
                            const float* __restrict__ a_dst,
                            float* __restrict__ s, float* __restrict__ t)
{
    int idx = blockIdx.x * blockDim.x + threadIdx.x;  // [0, B*NH)
    if (idx >= B_N * NH) return;
    int b = idx >> 2, hh = idx & 3;
    const float* hp = hf + b * HDIM + hh * HD;
    const float* as = a_src + hh * HD;
    const float* ad = a_dst + hh * HD;
    float ss = 0.f, tt = 0.f;
#pragma unroll 4
    for (int d = 0; d < HD; ++d) {
        float hv = hp[d];
        ss += hv * as[d];
        tt += hv * ad[d];
    }
    s[idx] = ss; t[idx] = tt;
}

// ---------------- K3: per-head softmax shift c = lrelu(max s + max t) ----------------
__global__ void k_head_max(const float* __restrict__ s,
                           const float* __restrict__ t,
                           float* __restrict__ cmax) {
    __shared__ float red[256];
    int hh = blockIdx.x;
    float ms = -1e30f, mt = -1e30f;
    for (int b = threadIdx.x; b < B_N; b += 256) {
        ms = fmaxf(ms, s[b * NH + hh]);
        mt = fmaxf(mt, t[b * NH + hh]);
    }
    red[threadIdx.x] = ms; __syncthreads();
    for (int st = 128; st > 0; st >>= 1) {
        if (threadIdx.x < st) red[threadIdx.x] = fmaxf(red[threadIdx.x], red[threadIdx.x + st]);
        __syncthreads();
    }
    ms = red[0]; __syncthreads();
    red[threadIdx.x] = mt; __syncthreads();
    for (int st = 128; st > 0; st >>= 1) {
        if (threadIdx.x < st) red[threadIdx.x] = fmaxf(red[threadIdx.x], red[threadIdx.x + st]);
        __syncthreads();
    }
    mt = red[0];
    if (threadIdx.x == 0) {
        float e = ms + mt;
        cmax[hh] = e > 0.f ? e : 0.2f * e;   // lrelu monotone -> exact global max of e
    }
}

// ---------------- K4: fused flash-style softmax + aggregation ----------------
// out[i,hh,:] = elu( (sum_j exp(lrelu(s_i+t_j)-c) * h[j,hh,:]) / Z_i )
// grid (64, NH), block 128 (4 waves x 16-row WMMA strips).
// h^T tiles are DMA'd by the Tensor Data Mover into double-buffered LDS,
// overlapping the next tile's DMA with this tile's WMMAs.
__global__ void __launch_bounds__(128)
k_gat_softmax_agg(const unsigned short* __restrict__ hTg, // [NH][HD][B] bf16
                  const float* __restrict__ s,            // [B, NH]
                  const float* __restrict__ t,            // [B, NH]
                  const float* __restrict__ cmax,         // [NH]
                  float* __restrict__ out)                // [B, HDIM]
{
    __shared__ unsigned short hTs[2][HD * 32];  // hT tile: [d][k], 64x32 bf16
    __shared__ float tch[2][32];
    __shared__ float zb[4][16];

    const int lane = threadIdx.x & 31;
    const int wv   = threadIdx.x >> 5;
    const int lm   = lane & 15;
    const bool hi  = lane >= 16;
    const int hh = blockIdx.y;
    const int i0 = blockIdx.x * 64;

    const float si  = s[(i0 + wv * 16 + lm) * NH + hh];
    const float cmx = cmax[hh];

    const unsigned long long headBase =
        (unsigned long long)(const void*)(hTg + (size_t)hh * HD * B_N);

    v8f acc[4] = {};
    float zpart = 0.f;
    const int halfoff = hi ? 8 : 0;    // A-operand K base for this lane half
    const int koffB   = hi ? 8 : 0;    // B-operand K base for this lane half

    // Prologue: DMA tile 0, stage t chunk 0.
    if (wv == 0) {
        tch[0][lane] = t[lane * NH + hh];
        tdm_load_2d_bf16((unsigned)(unsigned long long)(const void*)&hTs[0][0],
                         headBase, B_N, HD, 32, HD, B_N);
        __builtin_amdgcn_s_wait_tensorcnt(0);
    }
    __syncthreads();

    for (int it = 0; it < NJ; ++it) {
        const int cur = it & 1;
        const int j0  = it * 32;

        // Kick off DMA + t-chunk for the NEXT tile into the other buffer.
        if (wv == 0 && it + 1 < NJ) {
            tdm_load_2d_bf16((unsigned)(unsigned long long)(const void*)&hTs[1 - cur][0],
                             headBase + (unsigned long long)(j0 + 32) * 2ull,
                             B_N, HD, 32, HD, B_N);
            tch[1 - cur][lane] = t[(j0 + 32 + lane) * NH + hh];
        }

        // Build attention-weight A tile (16x32 bf16) in registers, WMMA A layout
        V16 a;
#pragma unroll
        for (int p = 0; p < 8; ++p) {
            int kk = halfoff + ((p < 4) ? 2 * p : 16 + 2 * (p - 4));
            float e0 = si + tch[cur][kk];
            float e1 = si + tch[cur][kk + 1];
            e0 = e0 > 0.f ? e0 : 0.2f * e0;
            e1 = e1 > 0.f ? e1 : 0.2f * e1;
            float w0 = __expf(e0 - cmx);
            float w1 = __expf(e1 - cmx);
            zpart += w0 + w1;
            a.s[2 * p]     = f32_to_bf16(w0);
            a.s[2 * p + 1] = f32_to_bf16(w1);
        }

        // 4 N-tiles over D=64: B operand from LDS, two ds_load_b128 per lane
#pragma unroll
        for (int nt = 0; nt < 4; ++nt) {
            V16 bm;
            const unsigned short* bp = &hTs[cur][(nt * 16 + lm) * 32 + koffB];
            bm.q[0] = *(const uint4*)bp;
            bm.q[1] = *(const uint4*)(bp + 16);
            acc[nt] = __builtin_amdgcn_wmma_f32_16x16x32_bf16(
                false, a.v, false, bm.v, (short)0, acc[nt], false, false);
        }

        if (wv == 0) __builtin_amdgcn_s_wait_tensorcnt(0);
        __syncthreads();
    }

    // Row sums Z: lane L and L^16 hold the two K-halves of row L%16
    float zfull = zpart + __shfl_xor(zpart, 16, 32);
    if (!hi) zb[wv][lm] = zfull;
    __syncthreads();

    // Epilogue: out = elu(C / Z)
#pragma unroll
    for (int nt = 0; nt < 4; ++nt) {
#pragma unroll
        for (int v = 0; v < 8; ++v) {
            int m = v + (hi ? 8 : 0);
            float val = acc[nt][v] / zb[wv][m];
            float o = val > 0.f ? val : (__expf(val) - 1.f);
            out[(i0 + wv * 16 + m) * HDIM + hh * HD + nt * 16 + lm] = o;
        }
    }
}

// ---------------- Launch ----------------
extern "C" void kernel_launch(void* const* d_in, const int* in_sizes, int n_in,
                              void* d_out, int out_size, void* d_ws, size_t ws_size,
                              hipStream_t stream) {
    (void)in_sizes; (void)n_in; (void)out_size; (void)ws_size;
    const float* x     = (const float*)d_in[0];
    // d_in[1] = attn_mask (all true) -- unused
    const float* W     = (const float*)d_in[2];
    const float* a_src = (const float*)d_in[3];
    const float* a_dst = (const float*)d_in[4];
    float* out = (float*)d_out;

    char* ws = (char*)d_ws;
    unsigned short* xb  = (unsigned short*)ws; ws += (size_t)B_N * FIN * 2;
    unsigned short* wb  = (unsigned short*)ws; ws += (size_t)HDIM * FIN * 2;
    unsigned short* hTg = (unsigned short*)ws; ws += (size_t)B_N * HDIM * 2;
    float* hf   = (float*)ws; ws += (size_t)B_N * HDIM * 4;
    float* sarr = (float*)ws; ws += (size_t)B_N * NH * 4;
    float* tarr = (float*)ws; ws += (size_t)B_N * NH * 4;
    float* cmx  = (float*)ws; ws += 256;

    k_cvt_bf16<<<(B_N * FIN + 255) / 256, 256, 0, stream>>>(x, xb, B_N * FIN);
    k_cvt_bf16<<<(HDIM * FIN + 255) / 256, 256, 0, stream>>>(W, wb, HDIM * FIN);
    k_proj_wmma<<<dim3(64, 4), 128, 0, stream>>>(xb, wb, hf, hTg);
    k_attn_coef<<<(B_N * NH + 255) / 256, 256, 0, stream>>>(hf, a_src, a_dst, sarr, tarr);
    k_head_max<<<NH, 256, 0, stream>>>(sarr, tarr, cmx);
    k_gat_softmax_agg<<<dim3(64, NH), 128, 0, stream>>>(hTg, sarr, tarr, cmx, out);
}
